// AttentionLayer_11184094839193
// MI455X (gfx1250) — matrix operand order repaired and checked by reading, not compiled, not run on hardware
//
#include <hip/hip_runtime.h>
#include <math.h>

// MI455X / gfx1250, wave32. bf16 WMMA flash-attention with LDS tile staging
// (async global->LDS copies, double buffered) to collapse per-wave redundant
// K/V traffic 8x within a workgroup.
typedef __attribute__((ext_vector_type(16))) __bf16 v16bf;
typedef __attribute__((ext_vector_type(8)))  __bf16 v8bf;
typedef __attribute__((ext_vector_type(8)))  float  v8f;
typedef __attribute__((ext_vector_type(4)))  int    v4i;

#define L_SZ 8192
#define D_SZ 256
#define S_SZ 256
#define LOG2E 1.4426950408889634f

#ifndef __has_builtin
#define __has_builtin(x) 0
#endif

#if defined(__HIP_DEVICE_COMPILE__) && __has_builtin(__builtin_amdgcn_global_load_async_to_lds_b128)
#define HAVE_ASYNC_LDS 1
#else
#define HAVE_ASYNC_LDS 0
#endif

#define AS1 __attribute__((address_space(1)))
#define AS3 __attribute__((address_space(3)))

// 16-byte global -> LDS copy; async (ASYNCcnt) when the gfx1250 builtin is
// available, otherwise a synchronous register bounce.
static __device__ __forceinline__ void cp16_to_lds(const __bf16* g, __bf16* l) {
#if HAVE_ASYNC_LDS
  __builtin_amdgcn_global_load_async_to_lds_b128((AS1 v4i*)g, (AS3 v4i*)l, 0, 0);
#else
  *(v8bf*)l = *(const v8bf*)g;
#endif
}
static __device__ __forceinline__ void async_wait_all() {
#if HAVE_ASYNC_LDS
  asm volatile("s_wait_asynccnt 0x0" ::: "memory");
#endif
}

static __device__ __forceinline__ v16bf combine16(v8bf lo, v8bf hi) {
  v16bf r;
#pragma unroll
  for (int i = 0; i < 8; ++i) { r[i] = lo[i]; r[i + 8] = hi[i]; }
  return r;
}

// ---------------------------------------------------------------------------
// Kernel 1: a = tanh(attended @ W_A_X) + b_A_X, plus derived flash inputs:
//   abf  = bf16(a)                         (K matrix, row major)
//   qbf  = bf16(2*log2e * W_A ⊙ a)         (Q matrix, row major, scale folded)
//   wsqL = log2e * sum_d W_A[d]*a[d]^2     (per-key softmax bias, log2 domain)
// ---------------------------------------------------------------------------
__global__ void prep_kernel(const float* __restrict__ attended,
                            const float* __restrict__ W_A_X,
                            const float* __restrict__ b_A_X,
                            const float* __restrict__ W_A,
                            __bf16* __restrict__ abf,
                            __bf16* __restrict__ qbf,
                            float* __restrict__ wsqL) {
  __shared__ float arow[D_SZ];
  __shared__ float red[D_SZ];
  const int i = blockIdx.x;
  const int t = threadIdx.x;
  arow[t] = attended[i * D_SZ + t];
  __syncthreads();
  float acc = 0.f;
#pragma unroll 8
  for (int k = 0; k < D_SZ; ++k) acc = fmaf(arow[k], W_A_X[k * D_SZ + t], acc);
  const float av = tanhf(acc) + b_A_X[t];
  const float wa = W_A[t];
  abf[i * D_SZ + t] = (__bf16)av;
  qbf[i * D_SZ + t] = (__bf16)(2.0f * LOG2E * wa * av);
  red[t] = wa * av * av;
  __syncthreads();
  for (int s = 128; s > 0; s >>= 1) {
    if (t < s) red[t] += red[t + s];
    __syncthreads();
  }
  if (t == 0) wsqL[i] = LOG2E * red[0];
}

// ---------------------------------------------------------------------------
// Kernel 1b: source[L,S] -> srcbf (bf16 row major) and srcT (bf16 [S, L]).
// ---------------------------------------------------------------------------
__global__ void src_prep_kernel(const float* __restrict__ source,
                                __bf16* __restrict__ srcbf,
                                __bf16* __restrict__ srcT) {
  __shared__ float tile[32][33];
  const int r0 = blockIdx.x * 32;
  const int c0 = blockIdx.y * 32;
  const int tx = threadIdx.x, ty = threadIdx.y;
#pragma unroll
  for (int j = 0; j < 32; j += 8) {
    float v = source[(r0 + ty + j) * S_SZ + c0 + tx];
    tile[ty + j][tx] = v;
    srcbf[(r0 + ty + j) * S_SZ + c0 + tx] = (__bf16)v;
  }
  __syncthreads();
#pragma unroll
  for (int j = 0; j < 32; j += 8) {
    srcT[(long)(c0 + ty + j) * L_SZ + r0 + tx] = (__bf16)tile[tx][ty + j];
  }
}

// Kernel 1c: W_A_combine [2S, S] -> WcT bf16 [S, 2S].
__global__ void wc_prep_kernel(const float* __restrict__ Wc,
                               __bf16* __restrict__ WcT) {
  const int idx = blockIdx.x * blockDim.x + threadIdx.x;
  const int k = idx >> 8, n = idx & 255;
  WcT[n * 512 + k] = (__bf16)Wc[idx];
}

// ---------------------------------------------------------------------------
// Kernel 2: flash attention. 8 waves/block share double-buffered K/V LDS
// tiles; one wave per 16 query rows; online softmax over 256 key-blocks of
// 32 keys. logits(log2) = Q·K^T - wsqL[key]; P = exp2(l - m).
// ---------------------------------------------------------------------------
__global__ void __launch_bounds__(256, 1)
flash_kernel(const __bf16* __restrict__ abf,
             const __bf16* __restrict__ qbf,
             const float* __restrict__ wsqL,
             const __bf16* __restrict__ srcT,
             __bf16* __restrict__ glim) {
  __shared__ __align__(32) __bf16 kbuf[2][32 * 256];  // 32 keys x D, row major
  __shared__ __align__(32) __bf16 vbuf[2][256 * 32];  // 256 cols x 32 keys
  __shared__ __align__(32) __bf16 pl[8][16 * 32];     // per-wave P tile
  const int tid = threadIdx.x;
  const int lane = tid & 31;
  const int w = tid >> 5;
  const int qb = blockIdx.x * 8 + w;     // query block 0..511
  const int half = (lane < 16) ? 0 : 1;
  const int l15 = lane & 15;

  // Cooperative stage of key-block kb into LDS buffer b (8 x 16B per thread).
  auto stage = [&](int kb, int b) {
    const int keybase = kb * 32;
    const __bf16* gk = abf + keybase * D_SZ;       // kbuf is a linear copy
#pragma unroll
    for (int i = 0; i < 4; ++i) {
      const int c = tid + i * 256;                 // chunk 0..1023
      cp16_to_lds(gk + c * 8, &kbuf[b][c * 8]);
    }
#pragma unroll
    for (int i = 0; i < 4; ++i) {
      const int c = tid + i * 256;                 // col = c/4, 4 chunks/col
      const int col = c >> 2, part = c & 3;
      cp16_to_lds(srcT + (long)col * L_SZ + keybase + part * 8,
                  &vbuf[b][c * 8]);
    }
  };

  stage(0, 0);

  // Preload Q A-operands for all of D=256 (8 x K=32 slices, 64 VGPRs).
  v16bf qa[8];
  {
    const __bf16* qrow = qbf + (qb * 16 + l15) * D_SZ;
#pragma unroll
    for (int ks = 0; ks < 8; ++ks) {
      const int d0 = ks * 32 + half * 16;
      v8bf lo = *(const v8bf*)(qrow + d0);
      v8bf hi = *(const v8bf*)(qrow + d0 + 8);
      qa[ks] = combine16(lo, hi);
    }
  }

  v8f O[16] = {};
  float mrow[8], lrow[8];
#pragma unroll
  for (int r = 0; r < 8; ++r) { mrow[r] = -3.0e38f; lrow[r] = 0.f; }

  async_wait_all();
  __syncthreads();

  const int NKB = L_SZ / 32;
  for (int kb = 0; kb < NKB; ++kb) {
    const int cur = kb & 1;
    if (kb + 1 < NKB) stage(kb + 1, cur ^ 1);      // prefetch next tile
    const int keybase = kb * 32;

    // ---- scores from LDS K tile: two 16x16 tiles over 32 keys ----
    v8f s0 = {}, s1 = {};
    {
      const __bf16* k0 = &kbuf[cur][l15 * D_SZ + half * 16];
      const __bf16* k1 = &kbuf[cur][(16 + l15) * D_SZ + half * 16];
#pragma unroll
      for (int ks = 0; ks < 8; ++ks) {
        v16bf b0 = *(const v16bf*)(k0 + ks * 32);
        s0 = __builtin_amdgcn_wmma_f32_16x16x32_bf16(false, qa[ks], false, b0,
                                                     (short)0, s0, false, false);
      }
#pragma unroll
      for (int ks = 0; ks < 8; ++ks) {
        v16bf b1 = *(const v16bf*)(k1 + ks * 32);
        s1 = __builtin_amdgcn_wmma_f32_16x16x32_bf16(false, qa[ks], false, b1,
                                                     (short)0, s1, false, false);
      }
    }
    const float w0 = wsqL[keybase + l15];
    const float w1 = wsqL[keybase + 16 + l15];

    // ---- online softmax (log2 domain) + stage P tile into LDS ----
    float scale[8];
#pragma unroll
    for (int r = 0; r < 8; ++r) {
      const float a0 = s0[r] - w0;
      const float a1 = s1[r] - w1;
      float tm = fmaxf(a0, a1);
      tm = fmaxf(tm, __shfl_xor(tm, 1, 32));
      tm = fmaxf(tm, __shfl_xor(tm, 2, 32));
      tm = fmaxf(tm, __shfl_xor(tm, 4, 32));
      tm = fmaxf(tm, __shfl_xor(tm, 8, 32));
      const float mn = fmaxf(mrow[r], tm);
      const float sc = __builtin_amdgcn_exp2f(mrow[r] - mn);
      mrow[r] = mn;
      const float p0 = __builtin_amdgcn_exp2f(a0 - mn);
      const float p1 = __builtin_amdgcn_exp2f(a1 - mn);
      float rs = p0 + p1;
      rs += __shfl_xor(rs, 1, 32);
      rs += __shfl_xor(rs, 2, 32);
      rs += __shfl_xor(rs, 4, 32);
      rs += __shfl_xor(rs, 8, 32);
      lrow[r] = lrow[r] * sc + rs;
      scale[r] = sc;
      const int row0 = half ? (r + 8) : r;
      pl[w][row0 * 32 + l15] = (__bf16)p0;
      pl[w][row0 * 32 + l15 + 16] = (__bf16)p1;
    }
#pragma unroll
    for (int nb = 0; nb < 16; ++nb) {
#pragma unroll
      for (int r = 0; r < 8; ++r) O[nb][r] *= scale[r];
    }

    // ---- P @ V from LDS V tile ----
    v8bf plo = *(const v8bf*)(&pl[w][l15 * 32 + half * 16]);
    v8bf phi = *(const v8bf*)(&pl[w][l15 * 32 + half * 16 + 8]);
    const v16bf pa = combine16(plo, phi);
#pragma unroll
    for (int nb = 0; nb < 16; ++nb) {
      v16bf bv = *(const v16bf*)(&vbuf[cur][(nb * 16 + l15) * 32 + half * 16]);
      O[nb] = __builtin_amdgcn_wmma_f32_16x16x32_bf16(false, pa, false, bv,
                                                      (short)0, O[nb], false, false);
    }

    async_wait_all();   // prefetched tile fully in LDS
    __syncthreads();    // ...and visible; everyone done reading buf[cur]
  }

  // ---- normalize and emit glimpsed (bf16, row major) ----
  float inv[8];
#pragma unroll
  for (int r = 0; r < 8; ++r) inv[r] = 1.0f / lrow[r];
#pragma unroll
  for (int nb = 0; nb < 16; ++nb) {
#pragma unroll
    for (int r = 0; r < 8; ++r) {
      const int row = qb * 16 + (half ? r + 8 : r);
      const int col = nb * 16 + l15;
      glim[row * S_SZ + col] = (__bf16)(O[nb][r] * inv[r]);
    }
  }
}

// ---------------------------------------------------------------------------
// Kernel 3: combined = tanh([glimpsed, source] @ W_A_combine) + b.
// ---------------------------------------------------------------------------
__global__ void __launch_bounds__(256, 1)
combine_kernel(const __bf16* __restrict__ glim,
               const __bf16* __restrict__ srcbf,
               const __bf16* __restrict__ WcT,
               const float* __restrict__ bc,
               float* __restrict__ out) {
  const int lane = threadIdx.x & 31;
  const int w = threadIdx.x >> 5;
  const int rb = blockIdx.x * 8 + w;
  const int half = (lane < 16) ? 0 : 1;
  const int l15 = lane & 15;

  v8f C[16] = {};
  for (int ks = 0; ks < 16; ++ks) {
    const __bf16* Abase = (ks < 8) ? glim : srcbf;
    const int kofs = (ks & 7) * 32;
    const __bf16* arow = Abase + (rb * 16 + l15) * 256 + kofs + half * 16;
    v8bf lo = *(const v8bf*)(arow);
    v8bf hi = *(const v8bf*)(arow + 8);
    const v16bf a = combine16(lo, hi);
    const __bf16* brow = WcT + ks * 32 + half * 16;
#pragma unroll
    for (int nb = 0; nb < 16; ++nb) {
      v16bf b = *(const v16bf*)(brow + (nb * 16 + l15) * 512);
      C[nb] = __builtin_amdgcn_wmma_f32_16x16x32_bf16(false, a, false, b,
                                                      (short)0, C[nb], false, false);
    }
  }
#pragma unroll
  for (int nb = 0; nb < 16; ++nb) {
    const int col = nb * 16 + l15;
    const float bias = bc[col];
#pragma unroll
    for (int r = 0; r < 8; ++r) {
      const int row = rb * 16 + (half ? r + 8 : r);
      out[row * 256 + col] = tanhf(C[nb][r]) + bias;
    }
  }
}

extern "C" void kernel_launch(void* const* d_in, const int* in_sizes, int n_in,
                              void* d_out, int out_size, void* d_ws, size_t ws_size,
                              hipStream_t stream) {
  const float* attended = (const float*)d_in[0];
  const float* source   = (const float*)d_in[1];
  const float* W_A_X    = (const float*)d_in[2];
  const float* b_A_X    = (const float*)d_in[3];
  const float* W_A      = (const float*)d_in[4];
  const float* W_A_comb = (const float*)d_in[5];
  const float* b_A_comb = (const float*)d_in[6];
  float* out = (float*)d_out;

  char* ws = (char*)d_ws;
  __bf16* abf   = (__bf16*)(ws + (0ull  << 20));  // 4 MB  K matrix bf16
  __bf16* qbf   = (__bf16*)(ws + (4ull  << 20));  // 4 MB  Q matrix bf16
  __bf16* srcT  = (__bf16*)(ws + (8ull  << 20));  // 4 MB  V transposed bf16
  __bf16* srcbf = (__bf16*)(ws + (12ull << 20));  // 4 MB  V row major bf16
  __bf16* glim  = (__bf16*)(ws + (16ull << 20));  // 4 MB  glimpsed bf16
  __bf16* WcT   = (__bf16*)(ws + (20ull << 20));  // 256 KB combine weights^T
  float*  wsqL  = (float*)(ws + (20ull << 20) + 512 * 256 * 2);  // 32 KB

  prep_kernel<<<L_SZ, D_SZ, 0, stream>>>(attended, W_A_X, b_A_X, W_A,
                                         abf, qbf, wsqL);
  src_prep_kernel<<<dim3(L_SZ / 32, S_SZ / 32), dim3(32, 8), 0, stream>>>(
      source, srcbf, srcT);
  wc_prep_kernel<<<(512 * 256) / 256, 256, 0, stream>>>(W_A_comb, WcT);
  flash_kernel<<<L_SZ / 16 / 8, 256, 0, stream>>>(abf, qbf, wsqL, srcT, glim);
  combine_kernel<<<L_SZ / 16 / 8, 256, 0, stream>>>(glim, srcbf, WcT,
                                                    b_A_comb, out);
}
